// CharNO_20607253087057
// MI455X (gfx1250) — matrix operand order repaired and checked by reading, not compiled, not run on hardware
//
#include <hip/hip_runtime.h>
#include <hip/hip_bf16.h>

// CDNA5 (gfx1250) wave32 WMMA implementation.
// One wave = one (b,q); 48 x v_wmma_f32_16x16x32_f16 per q (2 k-tiles x 3
// feature-chunks x 4 N-tiles x 2 MLPs). Seg-half of A and all layer-1 weights
// are register-resident across the per-wave q loop. Epilogue (exact GELU,
// sigmoid, masked softmin) is fully branch-free: FMA + v_exp_f32 + v_rcp_f32
// only, so EXEC stays all-ones through the whole loop (WMMA requirement) and
// the scheduler can interleave freely.

typedef __attribute__((ext_vector_type(16))) _Float16 v16h;
typedef __attribute__((ext_vector_type(8)))  float    v8f;

#define DEVI static __device__ __forceinline__

DEVI v8f wmma_f16(v16h a, v16h b, v8f c) {
    // (neg_a, A, neg_b, B, c_mod, C, reuse_a, reuse_b)
    return __builtin_amdgcn_wmma_f32_16x16x32_f16(false, a, false, b,
                                                  (short)0, c, false, false);
}

// Branch-free exact GELU: 0.5*x*(1+erf(x/sqrt(2))), erf via Abramowitz-Stegun
// 7.1.26 (|err| <= 1.5e-7), using hardware v_exp_f32 / v_rcp_f32.
DEVI float gelu_exact(float x) {
    const float z = __builtin_fabsf(x) * 0.70710678118654752f;
    const float t = __builtin_amdgcn_rcpf(__builtin_fmaf(0.3275911f, z, 1.0f));
    float p = __builtin_fmaf(1.061405429f, t, -1.453152027f);
    p = __builtin_fmaf(p, t, 1.421413741f);
    p = __builtin_fmaf(p, t, -0.284496736f);
    p = __builtin_fmaf(p, t, 0.254829592f);
    p *= t;
    const float e    = __expf(-z * z);                     // v_exp_f32
    const float erfa = __builtin_fmaf(-p, e, 1.0f);        // erf(|x|/sqrt2)
    const float er   = __builtin_copysignf(erfa, x);
    return 0.5f * x * (1.0f + er);
}

DEVI float sigmoid_f(float x) {
    return __builtin_amdgcn_rcpf(1.0f + __expf(-x));
}

// Pack 16 f32 from two runs of 8 (as 4 float4, pre-scaled) into WMMA A/B half layout.
DEVI v16h pack16(float4 a, float4 b, float4 c, float4 d, float s) {
    v16h t;
    t[0]=(_Float16)(a.x*s); t[1]=(_Float16)(a.y*s); t[2]=(_Float16)(a.z*s); t[3]=(_Float16)(a.w*s);
    t[4]=(_Float16)(b.x*s); t[5]=(_Float16)(b.y*s); t[6]=(_Float16)(b.z*s); t[7]=(_Float16)(b.w*s);
    t[8]=(_Float16)(c.x*s); t[9]=(_Float16)(c.y*s); t[10]=(_Float16)(c.z*s); t[11]=(_Float16)(c.w*s);
    t[12]=(_Float16)(d.x*s); t[13]=(_Float16)(d.y*s); t[14]=(_Float16)(d.z*s); t[15]=(_Float16)(d.w*s);
    return t;
}

__global__ __launch_bounds__(256, 1)
void charno_mlp_softmin_kernel(
    const float* __restrict__ seg,   // (B,K,H)   = (2,32,64)
    const float* __restrict__ cf,    // (B,Q,K,Ch)= (2,16384,32,32)
    const int*   __restrict__ pm,    // (B,K)
    const float* __restrict__ Ws1, const float* __restrict__ bs1,   // (96,64),(64)
    const float* __restrict__ Ws2, const float* __restrict__ bs2,   // (64,1),(1)
    const float* __restrict__ Wv1, const float* __restrict__ bv1,
    const float* __restrict__ Wv2, const float* __restrict__ bv2,
    const float* __restrict__ logT,
    float*       __restrict__ out)   // [0,32768): rho ; [32768, ...): weights
{
    constexpr int Kk = 32, H = 64, Q = 16384;

    const int lane = threadIdx.x & 31;
    const int lo   = lane & 15;        // column / row-in-tile selector
    const int hi   = lane >> 4;        // half-wave selector
    const int wave = (blockIdx.x * blockDim.x + threadIdx.x) >> 5;
    const int wavesPerB = (gridDim.x * blockDim.x) >> 6;   // total_waves / B
    const int b  = wave / wavesPerB;
    const int q0 = wave % wavesPerB;

    // ---------------- per-wave constant state (registers) ----------------
    // Layer-1 weights in WMMA B layout: Bm[mlp][chunk][ntile]; lane holds
    // column n = 16*nt + lo; VGPR v packs rows k = 2(v&3)+8*hi+16*(v>>2), +1.
    v16h Bm[2][3][4];
#pragma unroll
    for (int mlp = 0; mlp < 2; ++mlp) {
        const float* W1 = mlp ? Wv1 : Ws1;
#pragma unroll
        for (int c = 0; c < 3; ++c)
#pragma unroll
            for (int nt = 0; nt < 4; ++nt) {
                const int n = nt * 16 + lo;
                v16h t;
#pragma unroll
                for (int v = 0; v < 8; ++v) {
                    const int d0 = 32 * c + 2 * (v & 3) + 8 * hi + 16 * (v >> 2);
                    t[2*v]   = (_Float16)W1[(d0    ) * H + n];
                    t[2*v+1] = (_Float16)W1[(d0 + 1) * H + n];
                }
                Bm[mlp][c][nt] = t;
            }
    }
    // Layer-2 weights + layer-1 biases (per lane's 4 columns).
    float w2s[4], w2v[4], c0s[4], c0v[4];
#pragma unroll
    for (int nt = 0; nt < 4; ++nt) {
        const int n = nt * 16 + lo;
        w2s[nt] = Ws2[n]; w2v[nt] = Wv2[n];
        c0s[nt] = bs1[n]; c0v[nt] = bv1[n];
    }
    const float bias2s = bs2[0], bias2v = bv2[0];
    const float invT   = __expf(-logT[0]);   // 1/exp(log_temperature)

    // Piece masks for this lane's 16 k values (rows v+8*hi and 16+v+8*hi).
    float mskA[8], mskB[8];
#pragma unroll
    for (int v = 0; v < 8; ++v) {
        mskA[v] = (float)pm[b * Kk + v + 8 * hi];
        mskB[v] = (float)pm[b * Kk + 16 + v + 8 * hi];
    }

    // Seg half of A (feature dims 0..63), masked — constant over q.
    // A layout: lane row M = lo (tile k = lo+16*tile); half j -> K index
    // j + 8*hi (+8 extra for j>=8) => two contiguous runs of 8 floats.
    v16h Aseg[2][2];
#pragma unroll
    for (int tile = 0; tile < 2; ++tile) {
        const int k = lo + 16 * tile;
        const float mrow = (float)pm[b * Kk + k];
        const float* srow = seg + ((size_t)b * Kk + k) * H;
#pragma unroll
        for (int c = 0; c < 2; ++c) {
            const float* p = srow + 32 * c;
            float4 f0 = *(const float4*)(p + 8 * hi);
            float4 f1 = *(const float4*)(p + 8 * hi + 4);
            float4 f2 = *(const float4*)(p + 16 + 8 * hi);
            float4 f3 = *(const float4*)(p + 16 + 8 * hi + 4);
            Aseg[tile][c] = pack16(f0, f1, f2, f3, mrow);
        }
    }

    // ---------------- per-q loop (straight-line, EXEC all-ones) ----------
    for (int q = q0; q < Q; q += wavesPerB) {
        // Char half of A (feature dims 64..95).
        v16h Ach[2];
#pragma unroll
        for (int tile = 0; tile < 2; ++tile) {
            const int k = lo + 16 * tile;
            const float* crow = cf + (((size_t)b * Q + q) * Kk + k) * 32;
            float4 f0 = *(const float4*)(crow + 8 * hi);
            float4 f1 = *(const float4*)(crow + 8 * hi + 4);
            float4 f2 = *(const float4*)(crow + 16 + 8 * hi);
            float4 f3 = *(const float4*)(crow + 16 + 8 * hi + 4);
            Ach[tile] = pack16(f0, f1, f2, f3, 1.0f);
        }

        float sc[2][2][8];   // [mlp][tile][v] — reduced per-row outputs
#pragma unroll
        for (int mlp = 0; mlp < 2; ++mlp) {
#pragma unroll
            for (int tile = 0; tile < 2; ++tile) {
                v8f acc[4];
#pragma unroll
                for (int nt = 0; nt < 4; ++nt) {
                    const float bias = mlp ? c0v[nt] : c0s[nt];
                    v8f c;
#pragma unroll
                    for (int v = 0; v < 8; ++v) c[v] = bias;
                    c = wmma_f16(Aseg[tile][0], Bm[mlp][0][nt], c);
                    c = wmma_f16(Aseg[tile][1], Bm[mlp][1][nt], c);
                    c = wmma_f16(Ach[tile],     Bm[mlp][2][nt], c);
                    acc[nt] = c;
                }
                // GELU + layer-2 dot + 16-lane tree reduction per half.
#pragma unroll
                for (int v = 0; v < 8; ++v) {
                    float p = 0.0f;
#pragma unroll
                    for (int nt = 0; nt < 4; ++nt) {
                        const float w2 = mlp ? w2v[nt] : w2s[nt];
                        p = __builtin_fmaf(gelu_exact(acc[nt][v]), w2, p);
                    }
                    p += __shfl_xor(p, 1, 32);
                    p += __shfl_xor(p, 2, 32);
                    p += __shfl_xor(p, 4, 32);
                    p += __shfl_xor(p, 8, 32);
                    sc[mlp][tile][v] = p + (mlp ? bias2v : bias2s);
                }
            }
        }

        // Masked softmin over K=32 (softmax of -score/T), blended with sigmoid.
        float lg0[8], lg1[8], mx = -1e30f;
#pragma unroll
        for (int v = 0; v < 8; ++v) {
            lg0[v] = (mskA[v] > 0.5f) ? (-sc[0][0][v] * invT) : -1e30f;
            lg1[v] = (mskB[v] > 0.5f) ? (-sc[0][1][v] * invT) : -1e30f;
            mx = __builtin_fmaxf(mx, __builtin_fmaxf(lg0[v], lg1[v]));
        }
        mx = __builtin_fmaxf(mx, __shfl_xor(mx, 16, 32));

        float w0[8], w1[8], sum = 0.0f;
#pragma unroll
        for (int v = 0; v < 8; ++v) {
            w0[v] = mskA[v] * __expf(lg0[v] - mx);   // mask=0 -> weight 0
            w1[v] = mskB[v] * __expf(lg1[v] - mx);
            sum += w0[v] + w1[v];
        }
        sum += __shfl_xor(sum, 16, 32);
        const float rs = __builtin_amdgcn_rcpf(sum);

        float rho = 0.0f;
#pragma unroll
        for (int v = 0; v < 8; ++v) {
            w0[v] *= rs; w1[v] *= rs;
            rho = __builtin_fmaf(w0[v], sigmoid_f(sc[1][0][v]), rho);
            rho = __builtin_fmaf(w1[v], sigmoid_f(sc[1][1][v]), rho);
        }
        rho += __shfl_xor(rho, 16, 32);

        const size_t bq = (size_t)b * Q + q;
        if (lane == 0) out[bq] = rho;
        if (lo == 0) {  // lanes 0 and 16: each owns 16 of the 32 weights
            float* wout = out + 32768 + bq * 32 + 8 * hi;
            *(float4*)(wout)      = make_float4(w0[0], w0[1], w0[2], w0[3]);
            *(float4*)(wout + 4)  = make_float4(w0[4], w0[5], w0[6], w0[7]);
            *(float4*)(wout + 16) = make_float4(w1[0], w1[1], w1[2], w1[3]);
            *(float4*)(wout + 20) = make_float4(w1[4], w1[5], w1[6], w1[7]);
        }
    }
}

extern "C" void kernel_launch(void* const* d_in, const int* in_sizes, int n_in,
                              void* d_out, int out_size, void* d_ws, size_t ws_size,
                              hipStream_t stream) {
    (void)in_sizes; (void)n_in; (void)out_size; (void)d_ws; (void)ws_size;
    const float* seg  = (const float*)d_in[0];
    const float* cf   = (const float*)d_in[1];
    const int*   pm   = (const int*)  d_in[2];
    const float* Ws1  = (const float*)d_in[3];
    const float* bs1  = (const float*)d_in[4];
    const float* Ws2  = (const float*)d_in[5];
    const float* bs2  = (const float*)d_in[6];
    const float* Wv1  = (const float*)d_in[7];
    const float* bv1  = (const float*)d_in[8];
    const float* Wv2  = (const float*)d_in[9];
    const float* bv2  = (const float*)d_in[10];
    const float* logT = (const float*)d_in[11];

    // 512 blocks x 256 threads = 4096 wave32s; each wave owns one (b, q-stride)
    // lane-set and loops over 8 q values, keeping weights/seg-A in registers.
    charno_mlp_softmin_kernel<<<512, 256, 0, stream>>>(
        seg, cf, pm, Ws1, bs1, Ws2, bs2, Wv1, bv1, Wv2, bv2, logT,
        (float*)d_out);
}
